// _Block_58823872086489
// MI455X (gfx1250) — matrix operand order repaired and checked by reading, not compiled, not run on hardware
//
#include <hip/hip_runtime.h>
#include <hip/hip_bf16.h>

typedef __attribute__((ext_vector_type(16))) __bf16 v16bf;
typedef __attribute__((ext_vector_type(8)))  float  v8f;

#define DIM     768
#define NTOK    4096      // 2 * 2048
#define SEQ     2048
#define HEADS   12
#define HD      64
#define HIDDEN  3072
#define C3      2304      // 3 * DIM

__device__ __forceinline__ v8f v8f_zero() {
    v8f z = {0.f,0.f,0.f,0.f,0.f,0.f,0.f,0.f};
    return z;
}

__device__ __forceinline__ v8f wmma_bf16(v16bf a, v16bf b, v8f c) {
    // D = A(16x32 bf16) * B(32x16 bf16) + C(16x16 f32)
    return __builtin_amdgcn_wmma_f32_16x16x32_bf16(false, a, false, b, (short)0, c, false, false);
}

// CDNA5 async global->LDS copy, 16B per lane, tracked by ASYNCcnt (bypasses VGPRs).
__device__ __forceinline__ void async_load_b128(const void* gptr, void* lptr) {
    unsigned lds = (unsigned)(unsigned long long)lptr;   // flat low 32 bits == LDS byte offset
    unsigned long long ga = (unsigned long long)gptr;
    asm volatile("global_load_async_to_lds_b128 %0, %1, off"
                 :: "v"(lds), "v"(ga) : "memory");
}
__device__ __forceinline__ void wait_async0() {
    asm volatile("s_wait_asynccnt 0x0" ::: "memory");
}

// A-matrix fragment (16x32, bf16), row-major source: base -> element [0][0], stride in halves.
// ISA layout: lane m = lane&15; lanes<16: V0..3 k=0..7, V4..7 k=16..23; lanes>=16: +8.
__device__ __forceinline__ v16bf ld_frag_a(const __bf16* base, int stride) {
    int lane = threadIdx.x & 31;
    int m = lane & 15;
    int hb = (lane >> 4) << 3;                 // 0 or 8
    const __bf16* rp = base + m * stride;
    union { v16bf v; unsigned u[8]; } f;
#pragma unroll
    for (int i = 0; i < 8; ++i) {
        int k = ((i >> 2) << 4) + hb + ((i & 3) << 1);
        f.u[i] = *(const unsigned*)(rp + k);   // pair (k, k+1), 4B aligned
    }
    return f.v;
}

// B-matrix fragment (32x16, bf16). Source stored K-minor per column n (i.e. B^T [n][k]).
// ISA layout: lane n = lane&15; lanes<16: V0..3 k=0..7, V4..7 k=16..23; lanes>=16: +8.
__device__ __forceinline__ v16bf ld_frag_b(const __bf16* base, int stride) {
    int lane = threadIdx.x & 31;
    int n = lane & 15;
    int hb = (lane >> 4) << 3;                 // 0 or 8
    const __bf16* rp = base + n * stride;
    union { v16bf v; unsigned u[8]; } f;
#pragma unroll
    for (int i = 0; i < 8; ++i) {
        int k = ((i >> 2) << 4) + hb + ((i & 3) << 1);
        f.u[i] = *(const unsigned*)(rp + k);
    }
    return f.v;
}

// ---------------------------------------------------------------- f32 -> bf16
__global__ void f2bf_kernel(const float* __restrict__ in, __bf16* __restrict__ out, int n) {
    int i = blockIdx.x * 256 + threadIdx.x;
    if (i < n) out[i] = (__bf16)in[i];
}

// ---------------------------------------------------------------- LayerNorm -> bf16
__global__ __launch_bounds__(256)
void ln_bf16_kernel(const float* __restrict__ x, const float* __restrict__ g,
                    const float* __restrict__ bta, __bf16* __restrict__ out) {
    __shared__ float red[256];
    int row = blockIdx.x;
    int tid = threadIdx.x;
    const float* xr = x + (size_t)row * DIM;
    float xv[3];
    float s = 0.f;
#pragma unroll
    for (int i = 0; i < 3; ++i) { xv[i] = xr[tid + i * 256]; s += xv[i]; }
    red[tid] = s;
    __syncthreads();
    for (int off = 128; off > 0; off >>= 1) {
        if (tid < off) red[tid] += red[tid + off];
        __syncthreads();
    }
    float mu = red[0] * (1.f / DIM);
    __syncthreads();
    float vs = 0.f;
#pragma unroll
    for (int i = 0; i < 3; ++i) { float d = xv[i] - mu; vs += d * d; }
    red[tid] = vs;
    __syncthreads();
    for (int off = 128; off > 0; off >>= 1) {
        if (tid < off) red[tid] += red[tid + off];
        __syncthreads();
    }
    float rstd = rsqrtf(red[0] * (1.f / DIM) + 1e-5f);
#pragma unroll
    for (int i = 0; i < 3; ++i) {
        int c = tid + i * 256;
        out[(size_t)row * DIM + c] = (__bf16)((xv[i] - mu) * rstd * g[c] + bta[c]);
    }
}

// ---------------------------------------------------------------- tiled bf16 WMMA GEMM
// C[M,N] = A[M,K] @ B[K,N]  (+bias, +gelu, +residual), 128x128 block tile, BK=32,
// double-buffered LDS; A tile staged via async-to-LDS (layout-preserving),
// B tile staged transposed through VGPRs.
template <bool HAS_BIAS, bool GELU, bool RES, bool OUT_BF16>
__global__ __launch_bounds__(256)
void gemm_bf16_kernel(const __bf16* __restrict__ A, const __bf16* __restrict__ B,
                      const float* __restrict__ bias, const float* __restrict__ res,
                      float* __restrict__ outF, __bf16* __restrict__ outB,
                      int M, int N, int K) {
    constexpr int BM = 128, BN = 128, BK = 32, LSA = 40, LSB = 40;
    __shared__ __bf16 sA[2][BM * LSA];    // [m][k], row pitch 80B (16B aligned)
    __shared__ __bf16 sB[2][BN * LSB];    // B^T: [n][k]
    int tid  = threadIdx.x;
    int lane = tid & 31;
    int wave = tid >> 5;                   // 8 waves: 4 (M) x 2 (N)
    int bm = blockIdx.y * BM;
    int bn = blockIdx.x * BN;
    int wm = (wave >> 1) * 32;
    int wn = (wave & 1) * 64;

    auto stageA = [&](int buf, int k0) {   // 8KB via 2 async b128 per thread
#pragma unroll
        for (int i = 0; i < 2; ++i) {
            int c = tid + i * 256;         // 0..511 chunks of 16B
            int r  = c >> 2;               // 4 chunks per 64B row
            int co = (c & 3) << 3;         // half offset
            async_load_b128(&A[(size_t)(bm + r) * K + k0 + co],
                            &sA[buf][r * LSA + co]);
        }
    };
    auto stageB = [&](int buf, int k0) {   // transpose into [n][k]
#pragma unroll
        for (int i = 0; i < 16; ++i) {
            int idx = tid + i * 256;
            int kk = idx >> 7;
            int n  = idx & 127;
            sB[buf][n * LSB + kk] = B[(size_t)(k0 + kk) * N + bn + n];
        }
    };

    v8f acc[2][4];
#pragma unroll
    for (int i = 0; i < 2; ++i)
#pragma unroll
        for (int j = 0; j < 4; ++j) acc[i][j] = v8f_zero();

    stageA(0, 0);
    stageB(0, 0);
    wait_async0();
    __syncthreads();

    int nk = K / BK;
    int buf = 0;
    for (int kt = 0; kt < nk; ++kt) {
        if (kt + 1 < nk) {                 // prefetch next tile into the other buffer
            stageA(buf ^ 1, (kt + 1) * BK);
            stageB(buf ^ 1, (kt + 1) * BK);
        }
        v16bf af[2], bfr[4];
#pragma unroll
        for (int i = 0; i < 2; ++i) af[i] = ld_frag_a(sA[buf] + (wm + i * 16) * LSA, LSA);
#pragma unroll
        for (int j = 0; j < 4; ++j) bfr[j] = ld_frag_b(sB[buf] + (wn + j * 16) * LSB, LSB);
#pragma unroll
        for (int i = 0; i < 2; ++i)
#pragma unroll
            for (int j = 0; j < 4; ++j) acc[i][j] = wmma_bf16(af[i], bfr[j], acc[i][j]);
        wait_async0();
        __syncthreads();
        buf ^= 1;
    }

    // epilogue: C layout lane n = lane&15, row m = r + 8*(lane>=16)
    int cn = lane & 15;
    int rowadd = (lane >> 4) << 3;
#pragma unroll
    for (int i = 0; i < 2; ++i) {
#pragma unroll
        for (int j = 0; j < 4; ++j) {
            int gn = bn + wn + j * 16 + cn;
            float bv = HAS_BIAS ? bias[gn] : 0.f;
#pragma unroll
            for (int r = 0; r < 8; ++r) {
                int gm = bm + wm + i * 16 + r + rowadd;
                float v = acc[i][j][r] + bv;
                if (GELU) v = 0.5f * v * (1.f + erff(v * 0.70710678118654752f));
                if (RES)  v += res[(size_t)gm * N + gn];
                if (OUT_BF16) outB[(size_t)gm * N + gn] = (__bf16)v;
                else          outF[(size_t)gm * N + gn] = v;
            }
        }
    }
}

// ---------------------------------------------------------------- flash attention
// qkv: [NTOK][2304] bf16 (q @ h*64, k @ 768+h*64, v @ 1536+h*64) -> o: [NTOK][768] bf16
// Double-buffered K/V tiles; K staged via async-to-LDS, V staged transposed.
__global__ __launch_bounds__(128)
void attn_kernel(const __bf16* __restrict__ qkv, __bf16* __restrict__ o) {
    constexpr int KS = 72;   // K-tile LDS stride (halves), 144B rows (16B aligned)
    constexpr int VS = 36;   // V^T / P LDS stride (halves)
    __shared__ __bf16 sK[2][32 * KS];     // [key][dim]
    __shared__ __bf16 sVT[2][64 * VS];    // [dim][key]
    __shared__ __bf16 sP[4][16 * VS];     // per-wave P bounce [row][key]

    int b = blockIdx.z, h = blockIdx.y;
    int q0 = blockIdx.x * 64;
    int tid = threadIdx.x, lane = tid & 31, wave = tid >> 5;
    size_t tokbase = (size_t)b * SEQ;
    int qoff = h * HD, koff = DIM + h * HD, voff = 2 * DIM + h * HD;

    auto stageK = [&](int buf, int kbase) {   // 4KB via 2 async b128 per thread
#pragma unroll
        for (int i = 0; i < 2; ++i) {
            int c = tid + i * 128;            // 0..255 chunks of 16B
            int key = c >> 3;                 // 8 chunks per 128B row
            int co  = (c & 7) << 3;
            async_load_b128(&qkv[(tokbase + kbase + key) * C3 + koff + co],
                            &sK[buf][key * KS + co]);
        }
    };
    auto stageV = [&](int buf, int kbase) {   // transpose into [dim][key]
#pragma unroll
        for (int i = 0; i < 16; ++i) {
            int idx = tid + i * 128;
            int key = idx >> 6;
            int d   = idx & 63;
            sVT[buf][d * VS + key] = qkv[(tokbase + kbase + key) * C3 + voff + d];
        }
    };

    // Q fragments (16 rows x 64 dims -> two 16x32 A frags), loaded straight from global
    int qrow = q0 + wave * 16;
    v16bf aq[2];
#pragma unroll
    for (int c = 0; c < 2; ++c)
        aq[c] = ld_frag_a(qkv + (tokbase + qrow) * C3 + qoff + c * 32, C3);

    v8f accO[4];
#pragma unroll
    for (int d = 0; d < 4; ++d) accO[d] = v8f_zero();
    float mrow[8], lrow[8];
#pragma unroll
    for (int r = 0; r < 8; ++r) { mrow[r] = -1e30f; lrow[r] = 0.f; }

    int cn = lane & 15;
    int rowadd = (lane >> 4) << 3;

    stageK(0, 0);
    stageV(0, 0);
    wait_async0();
    __syncthreads();

    int buf = 0;
    for (int kt = 0; kt < SEQ / 32; ++kt) {
        if (kt + 1 < SEQ / 32) {
            stageK(buf ^ 1, (kt + 1) * 32);
            stageV(buf ^ 1, (kt + 1) * 32);
        }

        // S = Q @ K^T * scale : two 16x16 tiles over 32 keys
        v8f s[2];
#pragma unroll
        for (int nt = 0; nt < 2; ++nt) {
            v8f a = v8f_zero();
#pragma unroll
            for (int c = 0; c < 2; ++c)
                a = wmma_bf16(aq[c], ld_frag_b(sK[buf] + nt * 16 * KS + c * 32, KS), a);
#pragma unroll
            for (int r = 0; r < 8; ++r) a[r] *= 0.125f;   // hd^-0.5
            s[nt] = a;
        }

        // online softmax (row stats live replicated across each 16-lane half)
        float alpha[8], mnew[8];
#pragma unroll
        for (int r = 0; r < 8; ++r) {
            float v = fmaxf(s[0][r], s[1][r]);
            v = fmaxf(v, __shfl_xor(v, 1, 32));
            v = fmaxf(v, __shfl_xor(v, 2, 32));
            v = fmaxf(v, __shfl_xor(v, 4, 32));
            v = fmaxf(v, __shfl_xor(v, 8, 32));
            mnew[r]  = fmaxf(mrow[r], v);
            alpha[r] = __expf(mrow[r] - mnew[r]);
            mrow[r]  = mnew[r];
        }
#pragma unroll
        for (int r = 0; r < 8; ++r) {
            float p0 = __expf(s[0][r] - mnew[r]);
            float p1 = __expf(s[1][r] - mnew[r]);
            s[0][r] = p0; s[1][r] = p1;
            float t = p0 + p1;
            t += __shfl_xor(t, 1, 32);
            t += __shfl_xor(t, 2, 32);
            t += __shfl_xor(t, 4, 32);
            t += __shfl_xor(t, 8, 32);
            lrow[r] = lrow[r] * alpha[r] + t;
        }
#pragma unroll
        for (int d = 0; d < 4; ++d)
#pragma unroll
            for (int r = 0; r < 8; ++r) accO[d][r] *= alpha[r];

        // bounce P through per-wave LDS: C-layout -> A-fragment layout (same-wave DS is in-order)
        __bf16* pw = sP[wave];
#pragma unroll
        for (int nt = 0; nt < 2; ++nt)
#pragma unroll
            for (int r = 0; r < 8; ++r)
                pw[(r + rowadd) * VS + nt * 16 + cn] = (__bf16)s[nt][r];
        v16bf ap = ld_frag_a(pw, VS);

        // O += P @ V  (four 16x16 D tiles across head dim)
#pragma unroll
        for (int d = 0; d < 4; ++d)
            accO[d] = wmma_bf16(ap, ld_frag_b(sVT[buf] + d * 16 * VS, VS), accO[d]);

        wait_async0();
        __syncthreads();
        buf ^= 1;
    }

    // normalize and write merged-head output
#pragma unroll
    for (int d = 0; d < 4; ++d) {
        int gd = h * HD + d * 16 + cn;
#pragma unroll
        for (int r = 0; r < 8; ++r) {
            int gm = q0 + wave * 16 + r + rowadd;
            o[(tokbase + gm) * DIM + gd] = (__bf16)(accO[d][r] / lrow[r]);
        }
    }
}

// ---------------------------------------------------------------- launch
extern "C" void kernel_launch(void* const* d_in, const int* in_sizes, int n_in,
                              void* d_out, int out_size, void* d_ws, size_t ws_size,
                              hipStream_t stream) {
    const float* x      = (const float*)d_in[0];
    const float* ln1_g  = (const float*)d_in[1];
    const float* ln1_b  = (const float*)d_in[2];
    const float* w_qkv  = (const float*)d_in[3];
    const float* w_proj = (const float*)d_in[4];
    const float* b_proj = (const float*)d_in[5];
    const float* ln2_g  = (const float*)d_in[6];
    const float* ln2_b  = (const float*)d_in[7];
    const float* w_fc1  = (const float*)d_in[8];
    const float* b_fc1  = (const float*)d_in[9];
    const float* w_fc2  = (const float*)d_in[10];
    const float* b_fc2  = (const float*)d_in[11];
    float* out = (float*)d_out;

    char* w = (char*)d_ws;
    auto alloc = [&](size_t bytes) -> void* {
        void* p = (void*)w;
        w += (bytes + 255) & ~(size_t)255;
        return p;
    };
    __bf16* wqkv_bf  = (__bf16*)alloc((size_t)DIM * C3 * 2);
    __bf16* wproj_bf = (__bf16*)alloc((size_t)DIM * DIM * 2);
    __bf16* wfc1_bf  = (__bf16*)alloc((size_t)DIM * HIDDEN * 2);
    __bf16* wfc2_bf  = (__bf16*)alloc((size_t)HIDDEN * DIM * 2);
    __bf16* h1_bf    = (__bf16*)alloc((size_t)NTOK * DIM * 2);
    __bf16* qkv_bf   = (__bf16*)alloc((size_t)NTOK * C3 * 2);
    __bf16* o_bf     = (__bf16*)alloc((size_t)NTOK * DIM * 2);
    float*  x1_f     = (float*)alloc((size_t)NTOK * DIM * 4);
    __bf16* h2_bf    = (__bf16*)alloc((size_t)NTOK * DIM * 2);
    __bf16* m_bf     = (__bf16*)alloc((size_t)NTOK * HIDDEN * 2);

    auto cvt = [&](const float* src, __bf16* dst, int n) {
        f2bf_kernel<<<(n + 255) / 256, 256, 0, stream>>>(src, dst, n);
    };
    cvt(w_qkv,  wqkv_bf,  DIM * C3);
    cvt(w_proj, wproj_bf, DIM * DIM);
    cvt(w_fc1,  wfc1_bf,  DIM * HIDDEN);
    cvt(w_fc2,  wfc2_bf,  HIDDEN * DIM);

    // h1 = LN1(x)
    ln_bf16_kernel<<<NTOK, 256, 0, stream>>>(x, ln1_g, ln1_b, h1_bf);

    // qkv = h1 @ w_qkv
    gemm_bf16_kernel<false, false, false, true>
        <<<dim3(C3 / 128, NTOK / 128), 256, 0, stream>>>(
            h1_bf, wqkv_bf, nullptr, nullptr, nullptr, qkv_bf, NTOK, C3, DIM);

    // o = softmax(q k^T / sqrt(hd)) v  (flash, bf16 WMMA)
    attn_kernel<<<dim3(SEQ / 64, HEADS, 2), 128, 0, stream>>>(qkv_bf, o_bf);

    // x1 = x + o @ w_proj + b_proj
    gemm_bf16_kernel<true, false, true, false>
        <<<dim3(DIM / 128, NTOK / 128), 256, 0, stream>>>(
            o_bf, wproj_bf, b_proj, x, x1_f, nullptr, NTOK, DIM, DIM);

    // h2 = LN2(x1)
    ln_bf16_kernel<<<NTOK, 256, 0, stream>>>(x1_f, ln2_g, ln2_b, h2_bf);

    // m = gelu(h2 @ w_fc1 + b_fc1)
    gemm_bf16_kernel<true, true, false, true>
        <<<dim3(HIDDEN / 128, NTOK / 128), 256, 0, stream>>>(
            h2_bf, wfc1_bf, b_fc1, nullptr, nullptr, m_bf, NTOK, HIDDEN, DIM);

    // out = x1 + m @ w_fc2 + b_fc2
    gemm_bf16_kernel<true, false, true, false>
        <<<dim3(DIM / 128, NTOK / 128), 256, 0, stream>>>(
            m_bf, wfc2_bf, b_fc2, x1_f, out, nullptr, NTOK, DIM, HIDDEN);
}
